// VectorQuantizer_22033182228500
// MI455X (gfx1250) — matrix operand order repaired and checked by reading, not compiled, not run on hardware
//
#include <hip/hip_runtime.h>
#include <hip/hip_bf16.h>
#include <cfloat>

// ---- problem constants ----
#define BB   32
#define DD   256
#define HW   1024          // 32*32
#define NN   32768         // B*H*W
#define KK   2048
#define QELEMS 8388608     // B*D*H*W

#if __has_builtin(__builtin_amdgcn_global_load_async_to_lds_b128) && \
    __has_builtin(__builtin_amdgcn_s_wait_asynccnt)
#define USE_ASYNC 1
#define CT 32              // columns staged per barrier period (async path)
#else
#define USE_ASYNC 0
#define CT 16              // register-prefetch fallback (lower VGPR pressure)
#endif
#define NITER (KK / CT)
#define NGRP  (CT / 16)

typedef __attribute__((ext_vector_type(16))) __bf16 v16bf;
typedef __attribute__((ext_vector_type(8)))  __bf16 v8bf;
typedef __attribute__((ext_vector_type(8)))  float  v8f;
#if USE_ASYNC
// per the compiler diagnostic: params are vector_size(16) int pointers,
// global side in address_space(1), LDS side in address_space(3)
typedef __attribute__((__vector_size__(4 * sizeof(int)))) int v4i;
typedef __attribute__((address_space(1))) v4i gv4i;
typedef __attribute__((address_space(3))) v4i lv4i;
#endif

// ---------------------------------------------------------------------------
// 0) zero histogram + loss accumulator
// ---------------------------------------------------------------------------
__global__ void vq_init(int* counts, float* accum) {
    int t = blockIdx.x * blockDim.x + threadIdx.x;
    if (t < KK) counts[t] = 0;
    if (t == 0) accum[0] = 0.0f;
}

// ---------------------------------------------------------------------------
// 1) x [B,D,HW] -> A [N,D] bf16, via 32x32 LDS tile transpose (both sides
//    coalesced).  Row-normalization is provably unnecessary for the argmax.
// ---------------------------------------------------------------------------
__global__ __launch_bounds__(256) void vq_prep_a(const float* __restrict__ x,
                                                 __bf16* __restrict__ A) {
    __shared__ float tile[32][33];
    const int dt = blockIdx.x;   // D/32 = 8
    const int ht = blockIdx.y;   // HW/32 = 32
    const int b  = blockIdx.z;   // B = 32
    const int tx = threadIdx.x, ty = threadIdx.y;
    const size_t xb = (size_t)b * DD * HW;
#pragma unroll
    for (int j = 0; j < 4; ++j) {
        int d  = dt * 32 + ty + j * 8;
        int hw = ht * 32 + tx;
        tile[ty + j * 8][tx] = x[xb + (size_t)d * HW + hw];
    }
    __syncthreads();
#pragma unroll
    for (int j = 0; j < 4; ++j) {
        int hw = ht * 32 + ty + j * 8;
        int d  = dt * 32 + tx;
        A[((size_t)b * HW + hw) * DD + d] = (__bf16)tile[tx][ty + j * 8];
    }
}

// ---------------------------------------------------------------------------
// 2) table [D,K] -> Bt [K,D] bf16 (column-L2-normalized, for the GEMM)
//               -> rawT [K,D] f32 (unnormalized, for the coalesced gather)
// ---------------------------------------------------------------------------
__global__ __launch_bounds__(256) void vq_prep_b(const float* __restrict__ table,
                                                 __bf16* __restrict__ Bt,
                                                 float* __restrict__ rawT) {
    const int k = blockIdx.x;    // K blocks
    const int d = threadIdx.x;   // D threads
    float v = table[(size_t)d * KK + k];
    rawT[(size_t)k * DD + d] = v;
    __shared__ float red[256];
    red[d] = v * v;
    __syncthreads();
    for (int s = 128; s > 0; s >>= 1) {
        if (d < s) red[d] += red[d + s];
        __syncthreads();
    }
    __shared__ float scale;
    if (d == 0) scale = 1.0f / fmaxf(sqrtf(red[0]), 1e-12f);
    __syncthreads();
    Bt[(size_t)k * DD + d] = (__bf16)(v * scale);
}

// ---------------------------------------------------------------------------
// 3) bf16 WMMA GEMM with fused per-row argmax.
//    WG = 256 threads = 8 waves; each wave owns a 16-row tile (A resident in
//    64 VGPRs).  B tiles stream through LDS.
//
//    ASYNC path: GLOBAL_LOAD_ASYNC_TO_LDS moves the B tile global->LDS with
//    zero VGPR traffic (kills the register-prefetch spill seen at ~150 VGPR
//    pressure) and is tracked by ASYNCcnt.  Triple-buffered LDS makes ONE
//    barrier/iter sound: at iter t, s_wait_asynccnt(4) proves my tile-t group
//    landed (async loads complete in order; 4 issued per wave per tile); the
//    barrier proves everyone's tile-t landed and everyone finished
//    compute(t-1), freeing buffer (t-1)%3 == (t+2)%3, into which tile t+2 is
//    then issued.  Latency cover = two full compute iterations (32 WMMAs).
//
//    Fallback path (no async builtins): round-2 register double-buffer,
//    CT=16, single accumulator chain (~122 VGPRs, no spill).
// ---------------------------------------------------------------------------
__global__ __launch_bounds__(256) void vq_gemm_argmax(const __bf16* __restrict__ A,
                                                      const __bf16* __restrict__ Bt,
                                                      int* __restrict__ idxOut) {
#if USE_ASYNC
    __shared__ __align__(16) __bf16 ldsB[3][CT * DD];   // 3 x 16 KB
#else
    __shared__ __align__(16) __bf16 ldsB[2][CT * DD];   // 2 x 8 KB
#endif
    const int tid   = threadIdx.x;
    const int wave  = tid >> 5;
    const int lane  = tid & 31;
    const int laneM = lane & 15;
    const int half  = lane >> 4;
    const int nBase = blockIdx.x * 128 + wave * 16;

    // --- preload A fragments: lane holds row (lane&15); per 32-wide K slice,
    //     two 16B chunks at d0 + 8*half and d0 + 16 + 8*half (ISA A layout).
    v16bf afrag[8];
    const __bf16* arow = A + (size_t)(nBase + laneM) * DD;
#pragma unroll
    for (int f = 0; f < 8; ++f) {
        int d0 = f * 32 + half * 8;
        v8bf lo = *(const v8bf*)(arow + d0);
        v8bf hi = *(const v8bf*)(arow + d0 + 16);
#pragma unroll
        for (int i = 0; i < 8; ++i) { afrag[f][i] = lo[i]; afrag[f][8 + i] = hi[i]; }
    }

    float best[8];
    int   bidx[8];
#pragma unroll
    for (int rr = 0; rr < 8; ++rr) { best[rr] = -FLT_MAX; bidx[rr] = 0; }

#if USE_ASYNC
    // stage(tile, bsel): 4 async b128 per wave; 8 waves cover the 16 KB tile.
    auto stage = [&](int tile, int bsel) {
        const char* g = (const char*)(Bt + (size_t)tile * CT * DD);
        char*       l = (char*)ldsB[bsel];
#pragma unroll
        for (int j = 0; j < 4; ++j) {
            int off = (tid + j * 256) * 16;
            __builtin_amdgcn_global_load_async_to_lds_b128(
                (gv4i*)(g + off), (lv4i*)(l + off), 0, 0);
        }
    };
    stage(0, 0);
    stage(1, 1);

    int bsel = 0;
    for (int t = 0; t < NITER; ++t) {
        __builtin_amdgcn_s_wait_asynccnt(4);  // my tile-t group complete
        __syncthreads();                      // everyone's complete; (t-1)%3 freed
        {
            int tn = (t + 2) & (NITER - 1);   // branchless wrap at the tail
            int bn = bsel + 2; if (bn >= 3) bn -= 3;
            stage(tn, bn);
        }
        const __bf16* bufp = ldsB[bsel];
#else
    // --- register double-buffer fallback: prefetch tile 0 (32B per thread)
    uint4 r0, r1;
    {
        const uint4* src = (const uint4*)Bt;
        r0 = src[tid];
        r1 = src[tid + 256];
    }
    for (int t = 0; t < NITER; ++t) {
        const int buf = t & 1;
        uint4* dst = (uint4*)ldsB[buf];
        dst[tid]       = r0;
        dst[tid + 256] = r1;
        {   // issue next-tile loads BEFORE the barrier so they overlap compute
            const int tn = (t + 1) & (NITER - 1);
            const uint4* nsrc = (const uint4*)(Bt + (size_t)tn * CT * DD);
            r0 = nsrc[tid];
            r1 = nsrc[tid + 256];
        }
        __syncthreads();
        const __bf16* bufp = ldsB[buf];
#endif

#pragma unroll
        for (int g = 0; g < NGRP; ++g) {      // 16-column groups
            v8f c = {};
#pragma unroll
            for (int f = 0; f < 8; ++f) {
                // B layout: lane holds column (lane&15); two 8-wide K chunks
                // at d0 and d0 + 16, d0 = f*32 + 8*half (matches A).
                int d0 = f * 32 + half * 8;
                const __bf16* bp = bufp + (g * 16 + laneM) * DD + d0;
                v8bf lo = *(const v8bf*)(bp);
                v8bf hi = *(const v8bf*)(bp + 16);
                v16bf bfrag;
#pragma unroll
                for (int i = 0; i < 8; ++i) { bfrag[i] = lo[i]; bfrag[8 + i] = hi[i]; }
                c = __builtin_amdgcn_wmma_f32_16x16x32_bf16(
                        false, afrag[f], false, bfrag, (short)0, c, false, false);
            }
            // fused argmax: lane holds col; reg rr holds row rr + 8*half.
            // Strict > keeps the first (lowest) index.
            int col = t * CT + g * 16 + laneM;
#pragma unroll
            for (int rr = 0; rr < 8; ++rr) {
                float v = c[rr];
                if (v > best[rr]) { best[rr] = v; bidx[rr] = col; }
            }
        }
#if USE_ASYNC
        bsel += 1; if (bsel >= 3) bsel -= 3;
#endif
    }

    // reduce over the 16 lanes sharing each row (xor masks < 16 stay in-half)
#pragma unroll
    for (int m = 8; m >= 1; m >>= 1) {
#pragma unroll
        for (int rr = 0; rr < 8; ++rr) {
            float ov = __shfl_xor(best[rr], m, 32);
            int   oi = __shfl_xor(bidx[rr], m, 32);
            if (ov > best[rr] || (ov == best[rr] && oi < bidx[rr])) {
                best[rr] = ov; bidx[rr] = oi;
            }
        }
    }
    if (laneM == 0) {
#pragma unroll
        for (int rr = 0; rr < 8; ++rr)
            idxOut[nBase + half * 8 + rr] = bidx[rr];
    }
}

// ---------------------------------------------------------------------------
// 4) gather quantized output (coalesced store along HW, code rows contiguous
//    in rawT), accumulate sum((x-q)^2), histogram, and float index map.
//    One block per (b,d); histogram/index writes only from d==0 blocks.
// ---------------------------------------------------------------------------
__global__ __launch_bounds__(256) void vq_gather(const float* __restrict__ x,
                                                 const float* __restrict__ rawT,
                                                 const int* __restrict__ idx,
                                                 int* __restrict__ counts,
                                                 float* __restrict__ accum,
                                                 float* __restrict__ outQ,
                                                 float* __restrict__ outIdx) {
    const int bd = blockIdx.x;        // B*D = 8192
    const int b  = bd >> 8;
    const int d  = bd & 255;
    const int tid = threadIdx.x;
    const size_t baseX = (size_t)b * DD * HW + (size_t)d * HW;
    const int* idxRow = idx + b * HW;
    float lsum = 0.0f;
#pragma unroll
    for (int i = 0; i < 4; ++i) {
        int hw = tid + i * 256;
        int k  = idxRow[hw];
        float q  = rawT[(size_t)k * DD + d];
        float xv = x[baseX + hw];
        outQ[baseX + hw] = q;
        float df = xv - q;
        lsum += df * df;
        if (d == 0) {
            atomicAdd(&counts[k], 1);
            outIdx[b * HW + hw] = (float)k;
        }
    }
    __shared__ float red[256];
    red[tid] = lsum;
    __syncthreads();
    for (int s = 128; s > 0; s >>= 1) {
        if (tid < s) red[tid] += red[tid + s];
        __syncthreads();
    }
    if (tid == 0) atomicAdd(accum, red[0]);
}

// ---------------------------------------------------------------------------
// 5) scalars: dictionary_loss == commitment_loss == mean sq err (forward),
//    entropy = sum p*log(p+1e-10), p = counts/N.
// ---------------------------------------------------------------------------
__global__ __launch_bounds__(256) void vq_finalize(const int* __restrict__ counts,
                                                   const float* __restrict__ accum,
                                                   float* __restrict__ outScalars) {
    __shared__ float red[256];
    const int tid = threadIdx.x;
    float e = 0.0f;
    for (int k = tid; k < KK; k += 256) {
        float p = (float)counts[k] * (1.0f / (float)NN);
        e += p * logf(p + 1e-10f);
    }
    red[tid] = e;
    __syncthreads();
    for (int s = 128; s > 0; s >>= 1) {
        if (tid < s) red[tid] += red[tid + s];
        __syncthreads();
    }
    if (tid == 0) {
        float m = accum[0] * (1.0f / (float)QELEMS);
        outScalars[0] = m;        // dictionary_loss
        outScalars[1] = m;        // commitment_loss (same forward value)
        outScalars[2] = red[0];   // entropy_loss
    }
}

// ---------------------------------------------------------------------------
extern "C" void kernel_launch(void* const* d_in, const int* in_sizes, int n_in,
                              void* d_out, int out_size, void* d_ws, size_t ws_size,
                              hipStream_t stream) {
    const float* x     = (const float*)d_in[0];  // [B,D,H,W]
    const float* table = (const float*)d_in[1];  // [D,K]
    float* out = (float*)d_out;

    // workspace layout (16B aligned)
    char* ws = (char*)d_ws;
    __bf16* A    = (__bf16*)(ws);                        // 16 MB  [N,D] bf16
    __bf16* Bt   = (__bf16*)(ws + 16777216);             //  1 MB  [K,D] bf16 (normalized)
    float*  rawT = (float*) (ws + 17825792);             //  2 MB  [K,D] f32
    int*    idx  = (int*)   (ws + 19922944);             // 128 KB [N]
    int*    cnt  = (int*)   (ws + 20054016);             //   8 KB [K]
    float*  acc  = (float*) (ws + 20062208);             //   4 B

    vq_init<<<8, 256, 0, stream>>>(cnt, acc);
    vq_prep_a<<<dim3(8, 32, 32), dim3(32, 8, 1), 0, stream>>>(x, A);
    vq_prep_b<<<KK, 256, 0, stream>>>(table, Bt, rawT);
    vq_gemm_argmax<<<NN / 128, 256, 0, stream>>>(A, Bt, idx);
    vq_gather<<<BB * DD, 256, 0, stream>>>(x, rawT, idx, cnt, acc,
                                           out,                 // quantized [B,D,H,W]
                                           out + QELEMS + 3);   // indices as float
    vq_finalize<<<1, 256, 0, stream>>>(cnt, acc, out + QELEMS); // 3 scalars
}
